// Swin_Transformer_25228637897184
// MI455X (gfx1250) — matrix operand order repaired and checked by reading, not compiled, not run on hardware
//
#include <hip/hip_runtime.h>
#include <hip/hip_fp16.h>

typedef __attribute__((ext_vector_type(16))) _Float16 v16h;
typedef __attribute__((ext_vector_type(8)))  float    v8f;

#define DEV __device__ __forceinline__

// ---------------------------------------------------------------------------
// WMMA f32 = f16 x f16 + f32, 16x16x32 (CDNA5)
// ---------------------------------------------------------------------------
DEV v8f wmma16(v16h a, v16h b, v8f c) {
  return __builtin_amdgcn_wmma_f32_16x16x32_f16(
      /*neg_a=*/false, a, /*neg_b=*/false, b,
      /*c_mod=*/(short)0, c, /*reuse_a=*/false, /*reuse_b=*/false);
}

// Fragment-layout (swizzled) addressing. A fragment is 32 lanes x 16 halfs,
// stored contiguously (512 halfs = 1KB), 32B per lane -> 2x b128 per load.
//
// A-matrix element (m-row t, k-col c) (16-bit A 16x32), contiguous k order shared with B:
//   lane = (t&15) + 16*bit4(kk), e = kk&15, kk = c&31
DEV int swzA(int t, int c, int ksTot) {
  int frag = (t >> 4) * ksTot + (c >> 5);
  int kk = c & 31;
  int lane = (t & 15) + (((kk >> 4) & 1) << 4);
  int e = kk & 15;
  return (frag * 32 + lane) * 16 + e;
}
// B-matrix element (k, n) (16-bit B 32x16), contiguous k order as for A: lane = (n&15)+16*bit4(kk), e = kk&15
DEV int swzB(int k, int n, int nfTot) {
  int frag = (k >> 5) * nfTot + (n >> 4);
  int kk = k & 31;
  int lane = (n & 15) + (((kk >> 4) & 1) << 4);
  int e = kk & 15;
  return (frag * 32 + lane) * 16 + e;
}
// Load one fragment (LDS or global) from swizzled storage: 32B/lane
DEV v16h ldsw(const _Float16* p) {
  return *(const v16h*)(p + (threadIdx.x & 31) * 16);
}
DEV v16h ldgw(const _Float16* __restrict__ Wsw, int nfTot, int kf, int nf) {
  return *(const v16h*)(Wsw + (((long)(kf * nfTot + nf) * 32 + (threadIdx.x & 31)) * 16));
}

DEV int region256(int g) { return (g < 248) ? 0 : ((g < 252) ? 1 : 2); }

// ---------------------------------------------------------------------------
// Prepass: convert fp32 weight W[K][N] to f16 in WMMA B-fragment layout.
// One thread per (fragment, lane); writes 16 halfs (32B).
// ---------------------------------------------------------------------------
__global__ void __launch_bounds__(256)
swizzle_w_kernel(const float* __restrict__ W, int K, int N, _Float16* __restrict__ out) {
  int g = blockIdx.x * 256 + threadIdx.x;
  int total = (K >> 5) * (N >> 4) * 32;
  if (g >= total) return;
  int lane = g & 31;
  int frag = g >> 5;
  int nfTot = N >> 4;
  int kf = frag / nfTot, nf = frag - kf * nfTot;
  int n = nf * 16 + (lane & 15);
  long kb = (long)kf * 32 + ((lane >= 16) ? 16 : 0);
  v16h v;
#pragma unroll
  for (int e = 0; e < 16; ++e) v[e] = (_Float16)W[(kb + e) * N + n];
  *(v16h*)(out + (long)g * 16) = v;
}

// ---------------------------------------------------------------------------
// Attention kernel: 1 workgroup = 1 (8x8) window, 8 waves = 8 heads.
// out is [B,H,W,C] fp32 = shortcut + proj(attn-out)
// ---------------------------------------------------------------------------
__global__ void __launch_bounds__(256)
attn_kernel(const float* __restrict__ xin, int xin_nchw,
            const float* __restrict__ illu,
            const float* __restrict__ g1, const float* __restrict__ b1,
            const _Float16* __restrict__ qkvsw, const float* __restrict__ qkvb,
            const float* __restrict__ tbl,
            const _Float16* __restrict__ projsw, const float* __restrict__ pb,
            int shift, float* __restrict__ out) {
  const int tid  = threadIdx.x;
  const int lane = tid & 31;
  const int wave = tid >> 5;          // == head
  const bool hiw = lane >= 16;
  const int ln16 = lane & 15;

  const int blk  = blockIdx.x;
  const int bimg = blk >> 10;
  const int wy   = (blk >> 5) & 31;
  const int wx   = blk & 31;

  extern __shared__ char smem[];
  _Float16* qA_all = (_Float16*)smem;            // 8 * 2048 (A-sw, K padded 16->32)
  _Float16* kB_all = qA_all + 16384;             // 8 * 2048 (B-sw, K padded)
  _Float16* vB_all = kB_all + 16384;             // 8 * 1024 (B-sw, K=64,N=16)
  _Float16* sA_all = vB_all + 8192;              // 8 * 1024 (A-sw); overlaid iw[64][128]
  _Float16* xn     = sA_all + 8192;              // 8192 (A-sw); reused as ob
  float*    xraw   = (float*)(xn + 8192);        // [64][128] shortcut
  float*    tbl_s  = xraw + 8192;                // [225*8]
  float*    qkvb_s = tbl_s + 1800;               // [384]
  float*    pb_s   = qkvb_s + 384;               // [128]
  float*    mean_s = pb_s + 128;                 // [64]
  float*    rstd_s = mean_s + 64;                // [64]
  _Float16* qA = qA_all + wave * 2048;
  _Float16* kB = kB_all + wave * 2048;
  _Float16* vB = vB_all + wave * 1024;
  _Float16* sA = sA_all + wave * 1024;
  _Float16* iw = sA_all;                         // illumination window (phase 0/1 only)

  // ---- phase 0: params, window load (+roll), illu window, LayerNorm1 ----
  for (int i = tid; i < 1800; i += 256) tbl_s[i]  = tbl[i];
  for (int i = tid; i < 384;  i += 256) qkvb_s[i] = qkvb[i];
  for (int i = tid; i < 128;  i += 256) pb_s[i]   = pb[i];

  if (xin_nchw) {
    for (int e = tid; e < 8192; e += 256) {
      int c = e >> 6, t = e & 63;
      int oy = (wy * 8 + (t >> 3) + shift) & 255;
      int ox = (wx * 8 + (t & 7)  + shift) & 255;
      xraw[t * 128 + c] = xin[(((long)bimg * 128 + c) * 256 + oy) * 256 + ox];
    }
  } else {
    const float4* x4 = (const float4*)xin;
    for (int e = tid; e < 2048; e += 256) {
      int t = e >> 5, c4 = e & 31;
      int oy = (wy * 8 + (t >> 3) + shift) & 255;
      int ox = (wx * 8 + (t & 7)  + shift) & 255;
      ((float4*)xraw)[t * 32 + c4] = x4[(((long)bimg * 256 + oy) * 256 + ox) * 32 + c4];
    }
  }
  for (int e = tid; e < 8192; e += 256) {        // illu always NCHW; c-major coalesced
    int c = e >> 6, t = e & 63;
    int oy = (wy * 8 + (t >> 3) + shift) & 255;
    int ox = (wx * 8 + (t & 7)  + shift) & 255;
    iw[t * 128 + c] = (_Float16)illu[(((long)bimg * 128 + c) * 256 + oy) * 256 + ox];
  }
  __syncthreads();
  if (tid < 64) {
    float s = 0.f, s2 = 0.f;
    for (int c = 0; c < 128; ++c) { float v = xraw[tid * 128 + c]; s += v; s2 += v * v; }
    float m = s * (1.f / 128.f);
    float var = s2 * (1.f / 128.f) - m * m;
    mean_s[tid] = m;
    rstd_s[tid] = rsqrtf(var + 1e-5f);
  }
  __syncthreads();
  for (int e = tid; e < 8192; e += 256) {        // LN -> xn in A-fragment layout
    int t = e >> 7, c = e & 127;
    float v = (xraw[e] - mean_s[t]) * rstd_s[t] * g1[c] + b1[c];
    xn[swzA(t, c, 4)] = (_Float16)v;
  }
  __syncthreads();

  // ---- phase 1 (per wave/head): Q, K, V. 1 weight frag feeds 4 WMMAs ----
  const v8f z = {};
  for (int i = lane; i < 2048; i += 32) {        // zero K-dim padding regions
    qA[i] = (_Float16)0.f;
    kB[i] = (_Float16)0.f;
  }
  {  // Q (bias + scale)
    v8f acc[4];
#pragma unroll
    for (int m = 0; m < 4; ++m) acc[m] = z;
#pragma unroll
    for (int ks = 0; ks < 4; ++ks) {
      v16h b = ldgw(qkvsw, 24, ks, wave);
#pragma unroll
      for (int mt = 0; mt < 4; ++mt)
        acc[mt] = wmma16(ldsw(xn + (mt * 4 + ks) * 512), b, acc[mt]);
    }
    float qb_ = qkvb_s[wave * 16 + ln16];
#pragma unroll
    for (int mt = 0; mt < 4; ++mt)
#pragma unroll
      for (int r = 0; r < 8; ++r) {
        int row = mt * 16 + r + (hiw ? 8 : 0);
        qA[swzA(row, ln16, 1)] = (_Float16)((acc[mt][r] + qb_) * 0.25f);  // DH^-0.5
      }
  }
  {  // K -> B-layout transposed (d x token)
    v8f acc[4];
#pragma unroll
    for (int m = 0; m < 4; ++m) acc[m] = z;
#pragma unroll
    for (int ks = 0; ks < 4; ++ks) {
      v16h b = ldgw(qkvsw, 24, ks, 8 + wave);
#pragma unroll
      for (int mt = 0; mt < 4; ++mt)
        acc[mt] = wmma16(ldsw(xn + (mt * 4 + ks) * 512), b, acc[mt]);
    }
    float kb_ = qkvb_s[128 + wave * 16 + ln16];
#pragma unroll
    for (int mt = 0; mt < 4; ++mt)
#pragma unroll
      for (int r = 0; r < 8; ++r) {
        int tok = mt * 16 + r + (hiw ? 8 : 0);
        kB[swzB(ln16, tok, 4)] = (_Float16)(acc[mt][r] + kb_);
      }
  }
  {  // V, illumination-modulated, B-layout (token x d)
    v8f acc[4];
#pragma unroll
    for (int m = 0; m < 4; ++m) acc[m] = z;
#pragma unroll
    for (int ks = 0; ks < 4; ++ks) {
      v16h b = ldgw(qkvsw, 24, ks, 16 + wave);
#pragma unroll
      for (int mt = 0; mt < 4; ++mt)
        acc[mt] = wmma16(ldsw(xn + (mt * 4 + ks) * 512), b, acc[mt]);
    }
    float vb_ = qkvb_s[256 + wave * 16 + ln16];
#pragma unroll
    for (int mt = 0; mt < 4; ++mt)
#pragma unroll
      for (int r = 0; r < 8; ++r) {
        int t = mt * 16 + r + (hiw ? 8 : 0);
        float il = (float)iw[t * 128 + wave * 16 + ln16];
        vB[swzB(t, ln16, 1)] = (_Float16)((acc[mt][r] + vb_) * il);
      }
  }
  asm volatile("s_wait_dscnt 0" ::: "memory");
  __syncthreads();                               // xn free -> ob; iw free -> sA
  _Float16* ob = xn;

  // ---- phase 2: scores + bias + mask, softmax, P*V ----
#pragma unroll 1
  for (int mt = 0; mt < 4; ++mt) {
    v16h aq = ldsw(qA + mt * 512);
    v8f sc[4];
#pragma unroll
    for (int nt = 0; nt < 4; ++nt)
      sc[nt] = wmma16(aq, ldsw(kB + nt * 512), z);
    const int ti_base = mt * 16 + (hiw ? 8 : 0);
#pragma unroll
    for (int nt = 0; nt < 4; ++nt) {
      int tj  = nt * 16 + ln16;
      int tjy = tj >> 3, tjx = tj & 7;
      int cj  = shift ? (region256(wy * 8 + tjy) * 3 + region256(wx * 8 + tjx)) : 0;
#pragma unroll
      for (int r = 0; r < 8; ++r) {
        int ti  = ti_base + r;
        int tiy = ti >> 3, tix = ti & 7;
        int idx = (tiy - tjy + 7) * 15 + (tix - tjx + 7);
        float v = sc[nt][r] + tbl_s[idx * 8 + wave];
        if (shift) {
          int ci = region256(wy * 8 + tiy) * 3 + region256(wx * 8 + tix);
          if (ci != cj) v = -1e30f;
        }
        sc[nt][r] = v;
      }
    }
    // row softmax: each row lives in one 16-lane half (C/D layout)
#pragma unroll
    for (int r = 0; r < 8; ++r) {
      float m = fmaxf(fmaxf(sc[0][r], sc[1][r]), fmaxf(sc[2][r], sc[3][r]));
      m = fmaxf(m, __shfl_xor(m, 1, 16));
      m = fmaxf(m, __shfl_xor(m, 2, 16));
      m = fmaxf(m, __shfl_xor(m, 4, 16));
      m = fmaxf(m, __shfl_xor(m, 8, 16));
      float e0 = __expf(sc[0][r] - m), e1 = __expf(sc[1][r] - m);
      float e2 = __expf(sc[2][r] - m), e3 = __expf(sc[3][r] - m);
      float s = e0 + e1 + e2 + e3;
      s += __shfl_xor(s, 1, 16);
      s += __shfl_xor(s, 2, 16);
      s += __shfl_xor(s, 4, 16);
      s += __shfl_xor(s, 8, 16);
      float inv = 1.f / s;
      sc[0][r] = e0 * inv; sc[1][r] = e1 * inv;
      sc[2][r] = e2 * inv; sc[3][r] = e3 * inv;
    }
#pragma unroll
    for (int nt = 0; nt < 4; ++nt)
#pragma unroll
      for (int r = 0; r < 8; ++r)
        sA[swzA(r + (hiw ? 8 : 0), nt * 16 + ln16, 2)] = (_Float16)sc[nt][r];
    asm volatile("s_wait_dscnt 0" ::: "memory");
    v8f o = z;
    o = wmma16(ldsw(sA),       ldsw(vB),       o);
    o = wmma16(ldsw(sA + 512), ldsw(vB + 512), o);
#pragma unroll
    for (int r = 0; r < 8; ++r) {
      int t = mt * 16 + r + (hiw ? 8 : 0);
      ob[swzA(t, wave * 16 + ln16, 4)] = (_Float16)o[r];
    }
  }
  asm volatile("s_wait_dscnt 0" ::: "memory");
  __syncthreads();

  // ---- phase 3: proj (64x128 @ 128x128) + bias + shortcut, un-shift, store ----
  {
    v8f acc[4];
#pragma unroll
    for (int m = 0; m < 4; ++m) acc[m] = z;
#pragma unroll
    for (int ks = 0; ks < 4; ++ks) {
      v16h b = ldgw(projsw, 8, ks, wave);
#pragma unroll
      for (int mt = 0; mt < 4; ++mt)
        acc[mt] = wmma16(ldsw(ob + (mt * 4 + ks) * 512), b, acc[mt]);
    }
    int c = wave * 16 + ln16;
    float pbv = pb_s[c];
#pragma unroll
    for (int mt = 0; mt < 4; ++mt)
#pragma unroll
      for (int r = 0; r < 8; ++r) {
        int t  = mt * 16 + r + (hiw ? 8 : 0);
        int oy = (wy * 8 + (t >> 3) + shift) & 255;
        int ox = (wx * 8 + (t & 7)  + shift) & 255;
        out[(((long)bimg * 256 + oy) * 256 + ox) * 128 + c] =
            xraw[t * 128 + c] + acc[mt][r] + pbv;
      }
  }
}

// ---------------------------------------------------------------------------
// MLP kernel: 64 tokens / workgroup. y = x + fc2(gelu(fc1(LN2(x))))
// ---------------------------------------------------------------------------
__global__ void __launch_bounds__(256)
mlp_kernel(const float* __restrict__ xin,            // [B*H*W][128] NHWC
           const float* __restrict__ g2, const float* __restrict__ b2,
           const _Float16* __restrict__ f1sw, const float* __restrict__ f1b,
           const _Float16* __restrict__ f2sw, const float* __restrict__ f2b,
           int write_nchw, float* __restrict__ out) {
  const int tid  = threadIdx.x;
  const int lane = tid & 31;
  const int wave = tid >> 5;
  const bool hiw = lane >= 16;
  const int ln16 = lane & 15;
  const long base = (long)blockIdx.x * 64;

  extern __shared__ char smem[];
  _Float16* xn   = (_Float16*)smem;               // 8192 halfs, A-sw
  _Float16* hbuf = xn + 8192;                     // 32768 halfs, A-sw (K=512)
  float*    xraw = (float*)(hbuf + 32768);        // [64][128]
  float*    f1b_s = xraw + 8192;                  // [512]
  float*    f2b_s = f1b_s + 512;                  // [128]
  float*    mean_s = f2b_s + 128;                 // [64]
  float*    rstd_s = mean_s + 64;                 // [64]

  for (int i = tid; i < 512; i += 256) f1b_s[i] = f1b[i];
  for (int i = tid; i < 128; i += 256) f2b_s[i] = f2b[i];
  {
    const float4* x4 = (const float4*)(xin + base * 128);
    for (int e = tid; e < 2048; e += 256) ((float4*)xraw)[e] = x4[e];
  }
  __syncthreads();
  if (tid < 64) {
    float s = 0.f, s2 = 0.f;
    for (int c = 0; c < 128; ++c) { float v = xraw[tid * 128 + c]; s += v; s2 += v * v; }
    float m = s * (1.f / 128.f);
    float var = s2 * (1.f / 128.f) - m * m;
    mean_s[tid] = m;
    rstd_s[tid] = rsqrtf(var + 1e-5f);
  }
  __syncthreads();
  for (int e = tid; e < 8192; e += 256) {
    int t = e >> 7, c = e & 127;
    float v = (xraw[e] - mean_s[t]) * rstd_s[t] * g2[c] + b2[c];
    xn[swzA(t, c, 4)] = (_Float16)v;
  }
  __syncthreads();

  const v8f z = {};
  // fc1 (64x128 @ 128x512) + exact GELU -> hbuf (A-fragment layout, K=512)
#pragma unroll 1
  for (int nt = 0; nt < 4; ++nt) {
    int nf = wave * 4 + nt;
    v8f acc[4];
#pragma unroll
    for (int m = 0; m < 4; ++m) acc[m] = z;
#pragma unroll
    for (int ks = 0; ks < 4; ++ks) {
      v16h b = ldgw(f1sw, 32, ks, nf);
#pragma unroll
      for (int mt = 0; mt < 4; ++mt)
        acc[mt] = wmma16(ldsw(xn + (mt * 4 + ks) * 512), b, acc[mt]);
    }
    float bb = f1b_s[nf * 16 + ln16];
#pragma unroll
    for (int mt = 0; mt < 4; ++mt)
#pragma unroll
      for (int r = 0; r < 8; ++r) {
        float v = acc[mt][r] + bb;
        v = 0.5f * v * (1.f + erff(v * 0.70710678118f));
        int t = mt * 16 + r + (hiw ? 8 : 0);
        hbuf[swzA(t, nf * 16 + ln16, 16)] = (_Float16)v;
      }
  }
  asm volatile("s_wait_dscnt 0" ::: "memory");
  __syncthreads();

  // fc2 (64x512 @ 512x128) + bias + residual
  {
    v8f acc[4];
#pragma unroll
    for (int m = 0; m < 4; ++m) acc[m] = z;
#pragma unroll 1
    for (int ks = 0; ks < 16; ++ks) {
      v16h b = ldgw(f2sw, 8, ks, wave);
#pragma unroll
      for (int mt = 0; mt < 4; ++mt)
        acc[mt] = wmma16(ldsw(hbuf + (mt * 16 + ks) * 512), b, acc[mt]);
    }
    int c = wave * 16 + ln16;
    float bb = f2b_s[c];
#pragma unroll
    for (int mt = 0; mt < 4; ++mt)
#pragma unroll
      for (int r = 0; r < 8; ++r) {
        int t = mt * 16 + r + (hiw ? 8 : 0);
        float v = xraw[t * 128 + c] + acc[mt][r] + bb;
        long tg = base + t;
        if (write_nchw) {
          int bi = (int)(tg >> 16), rem = (int)(tg & 65535);
          int oy = rem >> 8, ox = rem & 255;
          out[(((long)bi * 128 + c) * 256 + oy) * 256 + ox] = v;
        } else {
          out[tg * 128 + c] = v;
        }
      }
  }
}

// ---------------------------------------------------------------------------
static constexpr int ATTN_SMEM = 114688 + 32768 + 7200 + 1536 + 512 + 256 + 256; // 157216
static constexpr int MLP_SMEM  = 16384 + 65536 + 32768 + 2048 + 512 + 256 + 256; // 117760

extern "C" void kernel_launch(void* const* d_in, const int* in_sizes, int n_in,
                              void* d_out, int out_size, void* d_ws, size_t ws_size,
                              hipStream_t stream) {
  (void)in_sizes; (void)n_in; (void)out_size; (void)ws_size;
  const float* x   = (const float*)d_in[0];
  const float* il  = (const float*)d_in[1];
  const float* g1  = (const float*)d_in[2];
  const float* b1  = (const float*)d_in[3];
  const float* qw  = (const float*)d_in[4];
  const float* qb  = (const float*)d_in[5];
  const float* tb  = (const float*)d_in[6];
  const float* pw  = (const float*)d_in[7];
  const float* pb  = (const float*)d_in[8];
  const float* g2  = (const float*)d_in[9];
  const float* b2  = (const float*)d_in[10];
  const float* f1w = (const float*)d_in[11];
  const float* f1b = (const float*)d_in[12];
  const float* f2w = (const float*)d_in[13];
  const float* f2b = (const float*)d_in[14];

  float* bufA = (float*)d_ws;    // attn-stage output [B,H,W,C], 134 MB
  float* bufX = (float*)d_out;   // block-0 staging (NHWC); final result (NCHW)

  // swizzled f16 weights live after bufA in the workspace
  _Float16* wsw    = (_Float16*)((char*)d_ws + 134217728);
  _Float16* qkvsw  = wsw;              // 2 * 49152 halfs
  _Float16* projsw = wsw + 98304;      // 2 * 16384
  _Float16* fc1sw  = wsw + 131072;     // 2 * 65536
  _Float16* fc2sw  = wsw + 262144;     // 2 * 65536

  hipFuncSetAttribute((const void*)attn_kernel,
                      hipFuncAttributeMaxDynamicSharedMemorySize, ATTN_SMEM);
  hipFuncSetAttribute((const void*)mlp_kernel,
                      hipFuncAttributeMaxDynamicSharedMemorySize, MLP_SMEM);

  auto sw = [&](const float* W, int K, int N, _Float16* o) {
    int total = (K >> 5) * (N >> 4) * 32;
    swizzle_w_kernel<<<dim3((total + 255) / 256), dim3(256), 0, stream>>>(W, K, N, o);
  };
  for (int i = 0; i < 2; ++i) {
    sw(qw  + (long)i * 128 * 384, 128, 384, qkvsw  + (long)i * 49152);
    sw(pw  + (long)i * 128 * 128, 128, 128, projsw + (long)i * 16384);
    sw(f1w + (long)i * 128 * 512, 128, 512, fc1sw  + (long)i * 65536);
    sw(f2w + (long)i * 512 * 128, 512, 128, fc2sw  + (long)i * 65536);
  }

  dim3 grid(4096), blkdim(256);
  // block 0: no shift, input NCHW
  attn_kernel<<<grid, blkdim, ATTN_SMEM, stream>>>(x, 1, il, g1, b1,
      qkvsw, qb, tb, projsw, pb, 0, bufA);
  mlp_kernel<<<grid, blkdim, MLP_SMEM, stream>>>(bufA, g2, b2,
      fc1sw, f1b, fc2sw, f2b, 0, bufX);
  // block 1: shift 4 + mask, input NHWC staging
  attn_kernel<<<grid, blkdim, ATTN_SMEM, stream>>>(bufX, 0, il,
      g1 + 128, b1 + 128, qkvsw + 49152, qb + 384, tb + 225 * 8,
      projsw + 16384, pb + 128, 4, bufA);
  mlp_kernel<<<grid, blkdim, MLP_SMEM, stream>>>(bufA,
      g2 + 128, b2 + 128, fc1sw + 65536, f1b + 512, fc2sw + 65536, f2b + 128,
      1, (float*)d_out);
}
